// Qwen35VLVisionModel_12146167513353
// MI455X (gfx1250) — compile-verified, matrix operand order checked by reading, not run on hardware
//
#include <hip/hip_runtime.h>

// ---------------------------------------------------------------------------
// Qwen3.5-VL vision blocks on MI455X (gfx1250, wave32, WMMA).
// bf16 data / f32 accumulate via V_WMMA_F32_16X16X32_BF16.
// Tile staging via GLOBAL_LOAD_ASYNC_TO_LDS_B128 (ASYNCcnt) + double buffering.
// Softmax lane reductions via DS_SWIZZLE_B32 (xor mode, 16-lane groups).
// ---------------------------------------------------------------------------

#define S_TOK   8192
#define DIM     1280
#define NHEADS  16
#define HDIM    80
#define HPAD    96          // 80 padded to 3 * 32 for K-steps
#define FF      5120
#define NSEG    8
#define SEGL    1024
#define DEPTH   2

typedef __attribute__((ext_vector_type(16))) __bf16 bf16x16;
typedef __attribute__((ext_vector_type(8)))  __bf16 bf16x8;
typedef __attribute__((ext_vector_type(8)))  float  f32x8;

union FragU { bf16x16 v; struct { bf16x8 lo; bf16x8 hi; } p; };

// A/B fragment load for 16x32 bf16 tile, row-major source (row stride in elems).
__device__ __forceinline__ bf16x16 frag_load(const __bf16* base, int stride, int k0, int lane) {
    int r = lane & 15;
    int h = (lane >> 4) * 8;
    const __bf16* p = base + (size_t)r * stride + k0 + h;
    FragU u;
    u.p.lo = *(const bf16x8*)(p);
    u.p.hi = *(const bf16x8*)(p + 16);
    return u.v;
}

__device__ __forceinline__ f32x8 wmma_bf16(bf16x16 a, bf16x16 b, f32x8 c) {
    return __builtin_amdgcn_wmma_f32_16x16x32_bf16(false, a, false, b, (short)0, c, false, false);
}

__device__ __forceinline__ float gelu_tanh(float v) {
    float u = 0.7978845608028654f * (v + 0.044715f * v * v * v);
    return 0.5f * v * (1.0f + tanhf(u));
}

// ds_swizzle xor-mode lane swap within 16-lane groups (single DS op, no memory).
#define SWZ(v, pat) __int_as_float(__builtin_amdgcn_ds_swizzle(__float_as_int(v), (pat)))

__device__ __forceinline__ float red_max16(float v) {
    v = fmaxf(v, SWZ(v, 0x041F));   // xor 1
    v = fmaxf(v, SWZ(v, 0x081F));   // xor 2
    v = fmaxf(v, SWZ(v, 0x101F));   // xor 4
    v = fmaxf(v, SWZ(v, 0x201F));   // xor 8
    return v;
}
__device__ __forceinline__ float red_sum16(float v) {
    v += SWZ(v, 0x041F);
    v += SWZ(v, 0x081F);
    v += SWZ(v, 0x101F);
    v += SWZ(v, 0x201F);
    return v;
}

// Issue one 16B async global->LDS copy (GVS mode: saddr base + 32-bit vaddr).
__device__ __forceinline__ void async_copy16(unsigned lds_addr, unsigned byte_off,
                                             const void* base) {
    asm volatile("global_load_async_to_lds_b128 %0, %1, %2"
                 :: "v"(lds_addr), "v"(byte_off), "s"(base) : "memory");
}
__device__ __forceinline__ void async_wait0() {
    asm volatile("s_wait_asynccnt 0x0" ::: "memory");
}

// ---------------------------------------------------------------------------
// float -> bf16 conversion (weights, one-shot)
// ---------------------------------------------------------------------------
__global__ __launch_bounds__(256) void f2bf_kernel(const float* __restrict__ in,
                                                   __bf16* __restrict__ out, size_t n) {
    size_t i = (size_t)blockIdx.x * 256 + threadIdx.x;
    size_t stride = (size_t)gridDim.x * 256;
    for (; i < n; i += stride) out[i] = (__bf16)in[i];
}

// ---------------------------------------------------------------------------
// LayerNorm: one block per token row (1280 elems), f32 in -> bf16 out
// ---------------------------------------------------------------------------
__global__ __launch_bounds__(256) void ln_kernel(const float* __restrict__ x,
                                                 const float* __restrict__ g,
                                                 const float* __restrict__ b,
                                                 __bf16* __restrict__ out) {
    __shared__ float red[256];
    int row = blockIdx.x, tid = threadIdx.x;
    const float* xr = x + (size_t)row * DIM;
    float v[5];
    float s = 0.f;
#pragma unroll
    for (int i = 0; i < 5; ++i) { v[i] = xr[tid + i * 256]; s += v[i]; }
    red[tid] = s; __syncthreads();
    for (int st = 128; st > 0; st >>= 1) { if (tid < st) red[tid] += red[tid + st]; __syncthreads(); }
    float mu = red[0] * (1.0f / DIM);
    __syncthreads();
    float q = 0.f;
#pragma unroll
    for (int i = 0; i < 5; ++i) { float d = v[i] - mu; q += d * d; }
    red[tid] = q; __syncthreads();
    for (int st = 128; st > 0; st >>= 1) { if (tid < st) red[tid] += red[tid + st]; __syncthreads(); }
    float inv = rsqrtf(red[0] * (1.0f / DIM) + 1e-6f);
#pragma unroll
    for (int i = 0; i < 5; ++i) {
        int c = tid + i * 256;
        out[(size_t)row * DIM + c] = (__bf16)((v[i] - mu) * inv * g[c] + b[c]);
    }
}

// ---------------------------------------------------------------------------
// GEMM: C[M,N] = act(A[M,K] x W[N,K]^T + bias) (+R)  -> outF (f32) / outB (bf16)
// Block tile 128x128, BK=64, 8 waves (4x2), wave tile 32x64 (2x4 WMMA frags).
// Double-buffered async-to-LDS staging: next K-tile streams in while WMMAs run.
// ---------------------------------------------------------------------------
template <int ACT, int HAS_R, int OUT_F, int OUT_B>
__global__ __launch_bounds__(256) void gemm_bf16_kernel(
    const __bf16* __restrict__ A, const __bf16* __restrict__ W,
    const float* __restrict__ bias, const float* __restrict__ R,
    float* __restrict__ outF, __bf16* __restrict__ outB,
    int N, int K) {
    __shared__ __bf16 At[2][128 * 64];
    __shared__ __bf16 Bt[2][128 * 64];
    int tid = threadIdx.x;
    int lane = tid & 31, w = tid >> 5;
    int wm = w >> 1, wn = w & 1;
    int m0 = blockIdx.y * 128, n0 = blockIdx.x * 128;

    auto issue_tile = [&](int kb, int buf) {
#pragma unroll
        for (int i = 0; i < 4; ++i) {             // 1024 16B-chunks / 256 thr / 2 mats
            int c = tid + i * 256;
            int row = c >> 3, col = (c & 7) * 8;
            unsigned la = (unsigned)(size_t)&At[buf][row * 64 + col];
            unsigned lb = (unsigned)(size_t)&Bt[buf][row * 64 + col];
            unsigned oa = ((unsigned)(m0 + row) * (unsigned)K + (unsigned)(kb + col)) * 2u;
            unsigned ob = ((unsigned)(n0 + row) * (unsigned)K + (unsigned)(kb + col)) * 2u;
            async_copy16(la, oa, A);
            async_copy16(lb, ob, W);
        }
    };

    f32x8 acc[2][4];
#pragma unroll
    for (int tm = 0; tm < 2; ++tm)
#pragma unroll
        for (int tn = 0; tn < 4; ++tn) acc[tm][tn] = (f32x8)(0.0f);

    issue_tile(0, 0);
    async_wait0();
    __syncthreads();

    int cur = 0;
    for (int kb = 0; kb < K; kb += 64) {
        if (kb + 64 < K) issue_tile(kb + 64, cur ^ 1);
#pragma unroll
        for (int kk = 0; kk < 2; ++kk) {
            bf16x16 af[2], bf[4];
#pragma unroll
            for (int tm = 0; tm < 2; ++tm)
                af[tm] = frag_load(&At[cur][(wm * 32 + tm * 16) * 64], 64, kk * 32, lane);
#pragma unroll
            for (int tn = 0; tn < 4; ++tn)
                bf[tn] = frag_load(&Bt[cur][(wn * 64 + tn * 16) * 64], 64, kk * 32, lane);
#pragma unroll
            for (int tm = 0; tm < 2; ++tm)
#pragma unroll
                for (int tn = 0; tn < 4; ++tn)
                    acc[tm][tn] = wmma_bf16(af[tm], bf[tn], acc[tm][tn]);
        }
        async_wait0();
        __syncthreads();
        cur ^= 1;
    }

    int r = lane & 15, half = lane >> 4;
#pragma unroll
    for (int tm = 0; tm < 2; ++tm)
#pragma unroll
        for (int tn = 0; tn < 4; ++tn) {
            int n = n0 + wn * 64 + tn * 16 + r;
            float bv = bias[n];
#pragma unroll
            for (int e = 0; e < 8; ++e) {
                int m = m0 + wm * 32 + tm * 16 + half * 8 + e;
                float v = acc[tm][tn][e] + bv;
                if (ACT) v = gelu_tanh(v);
                if (HAS_R) v += R[(size_t)m * N + n];
                if (OUT_F) outF[(size_t)m * N + n] = v;
                if (OUT_B) outB[(size_t)m * N + n] = (__bf16)v;
            }
        }
}

// ---------------------------------------------------------------------------
// RoPE + reshape: qkv f32 [S,3840] -> Q,K bf16 [seg*head][1024][96] (Q pre-scaled),
// V^T bf16 [seg*head][96][1024]. Pads d 80..95 with zeros.
// ---------------------------------------------------------------------------
__global__ __launch_bounds__(256) void rope_kernel(const float* __restrict__ qkv,
                                                   const float* __restrict__ cs,
                                                   const float* __restrict__ sn,
                                                   __bf16* __restrict__ Q,
                                                   __bf16* __restrict__ Kh,
                                                   __bf16* __restrict__ VT) {
    int s = blockIdx.x;
    int seg = s >> 10, n = s & 1023;
    const float scale = 0.11180339887498949f;  // 1/sqrt(80)
    for (int i = threadIdx.x; i < NHEADS * HPAD; i += 256) {
        int h = i / HPAD, d = i % HPAD;
        size_t dst = ((size_t)(seg * NHEADS + h) * SEGL + n) * HPAD + d;
        size_t qrow = (size_t)s * (3 * DIM);
        if (d >= HDIM) {
            Q[dst] = (__bf16)0.0f;
            Kh[dst] = (__bf16)0.0f;
        } else {
            float c = cs[(size_t)s * HDIM + d];
            float si = sn[(size_t)s * HDIM + d];
            int base = h * HDIM + d;
            int rotoff = (d < HDIM / 2) ? (HDIM / 2) : (-HDIM / 2);
            float rsign = (d < HDIM / 2) ? -1.f : 1.f;
            float qv = qkv[qrow + base];
            float qr = rsign * qkv[qrow + base + rotoff];
            Q[dst] = (__bf16)((qv * c + qr * si) * scale);
            float kv = qkv[qrow + DIM + base];
            float kr = rsign * qkv[qrow + DIM + base + rotoff];
            Kh[dst] = (__bf16)(kv * c + kr * si);
        }
        size_t dstv = ((size_t)(seg * NHEADS + h) * HPAD + d) * SEGL + n;
        VT[dstv] = (d < HDIM) ? (__bf16)qkv[qrow + 2 * DIM + h * HDIM + d] : (__bf16)0.0f;
    }
}

// ---------------------------------------------------------------------------
// Flash-style attention. Block = (seg*head, 128 query rows). 8 waves x 16 rows.
// K / V^T tiles streamed via async-to-LDS; online softmax with ds_swizzle
// reductions; P bounced via LDS into an A fragment for P x V.
// ---------------------------------------------------------------------------
__global__ __launch_bounds__(256) void attn_kernel(const __bf16* __restrict__ Q,
                                                   const __bf16* __restrict__ Kh,
                                                   const __bf16* __restrict__ VT,
                                                   __bf16* __restrict__ out) {
    __shared__ __bf16 Kt[32 * HPAD];       // 32 keys x 96
    __shared__ __bf16 Vt[HPAD * 32];       // V^T: 96 x 32 keys
    __shared__ __bf16 Pt[8][16 * 32];      // per-wave P tile
    int tid = threadIdx.x, lane = tid & 31, w = tid >> 5;
    int sh = blockIdx.x;                   // seg*16 + head
    int m0 = blockIdx.y * 128 + w * 16;    // query row within segment
    const __bf16* Qb = Q + ((size_t)sh * SEGL + m0) * HPAD;
    const __bf16* Kb = Kh + (size_t)sh * SEGL * HPAD;
    const __bf16* VTb = VT + (size_t)sh * HPAD * SEGL;

    bf16x16 aq[3];
#pragma unroll
    for (int kk = 0; kk < 3; ++kk) aq[kk] = frag_load(Qb, HPAD, kk * 32, lane);

    f32x8 o[5];
#pragma unroll
    for (int dt = 0; dt < 5; ++dt) o[dt] = (f32x8)(0.0f);
    float rmax[8], rsum[8];
#pragma unroll
    for (int r = 0; r < 8; ++r) { rmax[r] = -1e30f; rsum[r] = 0.f; }

    for (int n0 = 0; n0 < SEGL; n0 += 32) {
        __syncthreads();
        for (int c = tid; c < 384; c += 256) {         // 32*96/8 chunks
            int row = c / 12, col = (c % 12) * 8;
            unsigned ld = (unsigned)(size_t)&Kt[row * HPAD + col];
            unsigned off = ((unsigned)(n0 + row) * HPAD + (unsigned)col) * 2u;
            async_copy16(ld, off, Kb);
        }
        for (int c = tid; c < 384; c += 256) {         // 96*32/8 chunks
            int row = c / 4, col = (c % 4) * 8;
            unsigned ld = (unsigned)(size_t)&Vt[row * 32 + col];
            unsigned off = ((unsigned)row * SEGL + (unsigned)(n0 + col)) * 2u;
            async_copy16(ld, off, VTb);
        }
        async_wait0();
        __syncthreads();

        f32x8 s0 = (f32x8)(0.0f), s1 = (f32x8)(0.0f);
#pragma unroll
        for (int kk = 0; kk < 3; ++kk) {
            bf16x16 b0 = frag_load(&Kt[0], HPAD, kk * 32, lane);
            bf16x16 b1 = frag_load(&Kt[16 * HPAD], HPAD, kk * 32, lane);
            s0 = wmma_bf16(aq[kk], b0, s0);
            s1 = wmma_bf16(aq[kk], b1, s1);
        }

        int half = lane >> 4, col = lane & 15;
#pragma unroll
        for (int r = 0; r < 8; ++r) {
            float mx = red_max16(fmaxf(s0[r], s1[r]));
            float nm = fmaxf(rmax[r], mx);
            float sc = __expf(rmax[r] - nm);
            rmax[r] = nm;
            float p0 = __expf(s0[r] - nm);
            float p1 = __expf(s1[r] - nm);
            float ps = red_sum16(p0 + p1);
            rsum[r] = rsum[r] * sc + ps;
#pragma unroll
            for (int dt = 0; dt < 5; ++dt) o[dt][r] *= sc;
            int prow = r + half * 8;
            Pt[w][prow * 32 + col] = (__bf16)p0;
            Pt[w][prow * 32 + 16 + col] = (__bf16)p1;
        }
        asm volatile("s_wait_dscnt 0" ::: "memory");   // in-wave LDS bounce
        bf16x16 ap = frag_load(&Pt[w][0], 32, 0, lane);
#pragma unroll
        for (int dt = 0; dt < 5; ++dt) {
            bf16x16 bv = frag_load(&Vt[dt * 16 * 32], 32, 0, lane);
            o[dt] = wmma_bf16(ap, bv, o[dt]);
        }
    }

    int r = lane & 15, half = lane >> 4;
    int seg = sh >> 4, head = sh & 15;
#pragma unroll
    for (int dt = 0; dt < 5; ++dt) {
        int d = dt * 16 + r;
#pragma unroll
        for (int e = 0; e < 8; ++e) {
            int m = m0 + half * 8 + e;
            int tok = seg * SEGL + m;
            out[(size_t)tok * DIM + head * HDIM + d] = (__bf16)(o[dt][e] / rsum[e]);
        }
    }
}

// ---------------------------------------------------------------------------
// Host orchestration
// ---------------------------------------------------------------------------
extern "C" void kernel_launch(void* const* d_in, const int* in_sizes, int n_in,
                              void* d_out, int out_size, void* d_ws, size_t ws_size,
                              hipStream_t stream) {
    (void)in_sizes; (void)n_in; (void)out_size; (void)ws_size;
    const float* hidden = (const float*)d_in[0];
    const float* cosp   = (const float*)d_in[2];
    const float* sinp   = (const float*)d_in[3];
    const float* ln1g   = (const float*)d_in[4];
    const float* ln1b   = (const float*)d_in[5];
    const float* ln2g   = (const float*)d_in[6];
    const float* ln2b   = (const float*)d_in[7];
    const float* qkvw   = (const float*)d_in[8];
    const float* qkvb   = (const float*)d_in[9];
    const float* projw  = (const float*)d_in[10];
    const float* projb  = (const float*)d_in[11];
    const float* fc1w   = (const float*)d_in[12];
    const float* fc1b   = (const float*)d_in[13];
    const float* fc2w   = (const float*)d_in[14];
    const float* fc2b   = (const float*)d_in[15];

    char* p = (char*)d_ws;
    auto alloc = [&](size_t bytes) -> void* {
        void* r = (void*)p;
        p += (bytes + 255) & ~(size_t)255;
        return r;
    };
    float*  x       = (float*) alloc((size_t)S_TOK * DIM * 4);
    __bf16* h_bf    = (__bf16*)alloc((size_t)S_TOK * DIM * 2);
    float*  qkv_f   = (float*) alloc((size_t)S_TOK * 3 * DIM * 4);
    __bf16* Qb      = (__bf16*)alloc((size_t)NSEG * NHEADS * SEGL * HPAD * 2);
    __bf16* Kb      = (__bf16*)alloc((size_t)NSEG * NHEADS * SEGL * HPAD * 2);
    __bf16* VTb     = (__bf16*)alloc((size_t)NSEG * NHEADS * HPAD * SEGL * 2);
    __bf16* attn_bf = (__bf16*)alloc((size_t)S_TOK * DIM * 2);
    __bf16* g_bf    = (__bf16*)alloc((size_t)S_TOK * FF * 2);
    __bf16* qkvw_bf = (__bf16*)alloc((size_t)DEPTH * 3 * DIM * DIM * 2);
    __bf16* projw_bf= (__bf16*)alloc((size_t)DEPTH * DIM * DIM * 2);
    __bf16* fc1w_bf = (__bf16*)alloc((size_t)DEPTH * FF * DIM * 2);
    __bf16* fc2w_bf = (__bf16*)alloc((size_t)DEPTH * DIM * FF * 2);

    hipMemcpyAsync(x, hidden, (size_t)S_TOK * DIM * 4, hipMemcpyDeviceToDevice, stream);

    auto conv = [&](const float* src, __bf16* dst, size_t n) {
        int blocks = (int)((n + 255) / 256);
        if (blocks > 65535) blocks = 65535;
        f2bf_kernel<<<blocks, 256, 0, stream>>>(src, dst, n);
    };
    conv(qkvw,  qkvw_bf,  (size_t)DEPTH * 3 * DIM * DIM);
    conv(projw, projw_bf, (size_t)DEPTH * DIM * DIM);
    conv(fc1w,  fc1w_bf,  (size_t)DEPTH * FF * DIM);
    conv(fc2w,  fc2w_bf,  (size_t)DEPTH * DIM * FF);

    for (int i = 0; i < DEPTH; ++i) {
        // LN1 -> qkv
        ln_kernel<<<S_TOK, 256, 0, stream>>>(x, ln1g + i * DIM, ln1b + i * DIM, h_bf);
        gemm_bf16_kernel<0, 0, 1, 0><<<dim3(3 * DIM / 128, S_TOK / 128), 256, 0, stream>>>(
            h_bf, qkvw_bf + (size_t)i * 3 * DIM * DIM, qkvb + (size_t)i * 3 * DIM,
            nullptr, qkv_f, nullptr, 3 * DIM, DIM);
        // RoPE + layout
        rope_kernel<<<S_TOK, 256, 0, stream>>>(qkv_f, cosp, sinp, Qb, Kb, VTb);
        // attention
        attn_kernel<<<dim3(NSEG * NHEADS, SEGL / 128), 256, 0, stream>>>(Qb, Kb, VTb, attn_bf);
        // proj + residual (in-place on x)
        gemm_bf16_kernel<0, 1, 1, 0><<<dim3(DIM / 128, S_TOK / 128), 256, 0, stream>>>(
            attn_bf, projw_bf + (size_t)i * DIM * DIM, projb + (size_t)i * DIM,
            x, x, nullptr, DIM, DIM);
        // LN2 -> MLP
        ln_kernel<<<S_TOK, 256, 0, stream>>>(x, ln2g + i * DIM, ln2b + i * DIM, h_bf);
        gemm_bf16_kernel<1, 0, 0, 1><<<dim3(FF / 128, S_TOK / 128), 256, 0, stream>>>(
            h_bf, fc1w_bf + (size_t)i * FF * DIM, fc1b + (size_t)i * FF,
            nullptr, nullptr, g_bf, FF, DIM);
        gemm_bf16_kernel<0, 1, 1, 0><<<dim3(DIM / 128, S_TOK / 128), 256, 0, stream>>>(
            g_bf, fc2w_bf + (size_t)i * DIM * FF, fc2b + (size_t)i * DIM,
            x, x, nullptr, DIM, FF);
    }

    hipMemcpyAsync(d_out, x, (size_t)S_TOK * DIM * 4, hipMemcpyDeviceToDevice, stream);
}
